// TTTWrapper_38938173505802
// MI455X (gfx1250) — matrix operand order repaired and checked by reading, not compiled, not run on hardware
//
#include <hip/hip_runtime.h>
#include <math.h>

// ---------------------------------------------------------------------------
// MI455X (gfx1250) TTT pre-processing:
//   1) cvt_x / cvt_w : fp32 -> bf16 (W transposed+concatenated: WT[4096][1024])
//   2) gemm_wmma_bf16: Y[16384][4096] = Xbf16 @ WTbf16^T via v_wmma_f32_16x16x32_bf16
//      128x256x32 block tile, 8 waves, 64x64 wave tile = 16 WMMA/K-step/wave
//   3) lr_raw_kernel : lr_raw[16384][16] = X @ lr_weight^T + lr_bias
//   4) epilogue      : bias + l2norm + RoPE + V-layernorm + eta(sigmoid) +
//                      interleave permutation, written straight into d_out.
// ---------------------------------------------------------------------------

typedef __bf16 bf16;
typedef __attribute__((ext_vector_type(16))) __bf16 v16bf;
typedef __attribute__((ext_vector_type(8)))  __bf16 v8bf;
typedef __attribute__((ext_vector_type(8)))  float  v8f;

#define B_    2
#define L_    8192
#define D_    1024
#define H_    16
#define HD_   64
#define NC_   128
#define MINI_ 64
#define NBIG  4096          // 4*D output columns of fused GEMM
#define ROWS  16384         // B*L

#define BM 128
#define BN 256
#define BK 32
#define LDT 40              // padded LDS stride in bf16 (80B, 16B aligned, conflict-free)

// interleave: output seq position -> source seq position (see reference _interleave)
__device__ __forceinline__ int src_pos(int o) {
    int ch = o >> 11;              // / 2048
    int r  = o & 2047;
    if (r < 64) return ch * 64 + r;           // text piece
    return 256 + ch * 1984 + (r - 64);        // video piece
}

__device__ __forceinline__ float wave_sum(float x) {
#pragma unroll
    for (int m = 16; m > 0; m >>= 1) x += __shfl_xor(x, m, 32);
    return x;
}

// ---------------------------------------------------------------------------
__global__ void cvt_x_kernel(const float* __restrict__ X, bf16* __restrict__ Xb) {
    size_t idx = ((size_t)blockIdx.x * blockDim.x + threadIdx.x) * 8;
    float4 a = *(const float4*)(X + idx);
    float4 c = *(const float4*)(X + idx + 4);
    v8bf o;
    o[0] = (bf16)a.x; o[1] = (bf16)a.y; o[2] = (bf16)a.z; o[3] = (bf16)a.w;
    o[4] = (bf16)c.x; o[5] = (bf16)c.y; o[6] = (bf16)c.z; o[7] = (bf16)c.w;
    *(v8bf*)(Xb + idx) = o;
}

// WT[n][d] = W_{n/1024}[d][n%1024]   (transposed so B fragments load contiguously)
__global__ void cvt_w_kernel(const float* __restrict__ wq, const float* __restrict__ wk,
                             const float* __restrict__ wv, const float* __restrict__ wu,
                             bf16* __restrict__ WT) {
    size_t gid = (size_t)blockIdx.x * blockDim.x + threadIdx.x;  // 0 .. 4096*1024-1
    int n = (int)(gid >> 10);
    int d = (int)(gid & 1023);
    int which = n >> 10;
    int c = n & 1023;
    const float* w = (which == 0) ? wq : (which == 1) ? wk : (which == 2) ? wv : wu;
    WT[gid] = (bf16)w[(size_t)d * D_ + c];
}

// ---------------------------------------------------------------------------
__global__ __launch_bounds__(256) void gemm_wmma_bf16(
    const bf16* __restrict__ A, const bf16* __restrict__ Bt, float* __restrict__ Y) {
    __shared__ bf16 As[2][BM * LDT];   // 2 x 10240 B
    __shared__ bf16 Bs[2][BN * LDT];   // 2 x 20480 B   (total 60 KB)

    const int tid   = threadIdx.x;
    const int lane  = tid & 31;
    const int wid   = tid >> 5;
    const int wm    = wid >> 2;        // 0..1  (64 rows each)
    const int wn    = wid & 3;         // 0..3  (64 cols each)
    const int tileM = blockIdx.y * BM;
    const int tileN = blockIdx.x * BN;

    // cooperative loads: A -> thread (row=tid/2, 16-elem seg), B -> thread = full 32-elem row
    const int arow = tid >> 1;         // 0..127
    const int aseg = tid & 1;          // 0/1
    const bf16* agp = A  + (size_t)(tileM + arow) * D_ + aseg * 16;
    const bf16* bgp = Bt + (size_t)(tileN + tid) * D_;

    v8f acc[4][4];
    {
        v8f zf = {};
#pragma unroll
        for (int m = 0; m < 4; ++m)
#pragma unroll
            for (int n = 0; n < 4; ++n) acc[m][n] = zf;
    }

    // stage 0
    {
        uint4 a0 = *(const uint4*)(agp);
        uint4 a1 = *(const uint4*)(agp + 8);
        bf16* as = &As[0][arow * LDT + aseg * 16];
        *(uint4*)(as) = a0; *(uint4*)(as + 8) = a1;
        bf16* bs = &Bs[0][tid * LDT];
#pragma unroll
        for (int j = 0; j < 4; ++j)
            *(uint4*)(bs + j * 8) = *(const uint4*)(bgp + j * 8);
    }
    __syncthreads();

    const int half = lane >> 4;        // K-half select (ISA 16-bit A layout)
    const int rc   = lane & 15;        // M (A) / N (B) index within fragment
    const int NT   = D_ / BK;          // 32 K-steps

    for (int kt = 0; kt < NT; ++kt) {
        const int buf = kt & 1;
        uint4 a0, a1, b0, b1, b2, b3;
        if (kt + 1 < NT) {
            const bf16* ap = agp + (kt + 1) * BK;
            const bf16* bp = bgp + (kt + 1) * BK;
            a0 = *(const uint4*)(ap);      a1 = *(const uint4*)(ap + 8);
            b0 = *(const uint4*)(bp);      b1 = *(const uint4*)(bp + 8);
            b2 = *(const uint4*)(bp + 16); b3 = *(const uint4*)(bp + 24);
            if (kt + 2 < NT) {  // gfx1250 global_prefetch_b8 for the tile after next
                __builtin_prefetch(agp + (kt + 2) * BK, 0, 1);
                __builtin_prefetch(bgp + (kt + 2) * BK, 0, 1);
            }
        }

        v16bf afrag[4], bfrag[4];
#pragma unroll
        for (int m = 0; m < 4; ++m) {
            // lanes 0-15: K 0..7 & 16..23 ; lanes 16-31: K 8..15 & 24..31
            const bf16* p = &As[buf][(wm * 64 + m * 16 + rc) * LDT + half * 8];
            union { v16bf v; uint4 u[2]; } t;
            t.u[0] = *(const uint4*)(p);
            t.u[1] = *(const uint4*)(p + 16);
            afrag[m] = t.v;
        }
#pragma unroll
        for (int n = 0; n < 4; ++n) {
            // lanes 0-15: K 0..7 & 16..23 ; lanes 16-31: K 8..15 & 24..31
            const bf16* p = &Bs[buf][(wn * 64 + n * 16 + rc) * LDT + half * 8];
            union { v16bf v; uint4 u[2]; } t;
            t.u[0] = *(const uint4*)(p);
            t.u[1] = *(const uint4*)(p + 16);
            bfrag[n] = t.v;
        }

#pragma unroll
        for (int m = 0; m < 4; ++m)
#pragma unroll
            for (int n = 0; n < 4; ++n)
                acc[m][n] = __builtin_amdgcn_wmma_f32_16x16x32_bf16(
                    false, afrag[m], false, bfrag[n], (short)0, acc[m][n], false, false);

        __syncthreads();
        if (kt + 1 < NT) {
            bf16* as = &As[buf ^ 1][arow * LDT + aseg * 16];
            *(uint4*)(as) = a0; *(uint4*)(as + 8) = a1;
            bf16* bs = &Bs[buf ^ 1][tid * LDT];
            *(uint4*)(bs)      = b0; *(uint4*)(bs + 8)  = b1;
            *(uint4*)(bs + 16) = b2; *(uint4*)(bs + 24) = b3;
        }
        __syncthreads();
    }

    // C/D layout: VGPR r -> M = r + 8*(lane>=16), N = lane%16
#pragma unroll
    for (int m = 0; m < 4; ++m)
#pragma unroll
        for (int n = 0; n < 4; ++n) {
            const int col  = tileN + wn * 64 + n * 16 + rc;
            const int row0 = tileM + wm * 64 + m * 16 + (half << 3);
            float* yp = Y + (size_t)row0 * NBIG + col;
#pragma unroll
            for (int r = 0; r < 8; ++r) yp[(size_t)r * NBIG] = acc[m][n][r];
        }
}

// ---------------------------------------------------------------------------
// lr_raw[row][h] = dot(X[row,:], lr_weight[h,0,:]) + lr_bias[h]
__global__ __launch_bounds__(256) void lr_raw_kernel(
    const float* __restrict__ X, const float* __restrict__ lrw,
    const float* __restrict__ lrb, float* __restrict__ outp) {
    __shared__ float xs[D_];
    const int row = blockIdx.x;
    const float* xr = X + (size_t)row * D_;
    for (int i = threadIdx.x; i < D_; i += 256) xs[i] = xr[i];
    __syncthreads();

    const int h    = threadIdx.x >> 4;   // 0..15
    const int part = threadIdx.x & 15;   // 0..15 (stays inside one wave half)
    const float* w = lrw + (size_t)h * D_;
    float s = 0.f;
    for (int c = part; c < D_; c += 16) s += xs[c] * w[c];
#pragma unroll
    for (int off = 8; off > 0; off >>= 1) s += __shfl_xor(s, off, 32);
    if (part == 0) outp[(size_t)row * H_ + h] = s + lrb[h];
}

// ---------------------------------------------------------------------------
// One wave32 per (b, h, output-seq) 64-element row: 2 elems/lane.
__global__ __launch_bounds__(256) void epilogue_kernel(
    const float* __restrict__ Y, const float* __restrict__ freqs,
    const float* __restrict__ qb, const float* __restrict__ kb,
    const float* __restrict__ vb, const float* __restrict__ ub,
    const float* __restrict__ nw, const float* __restrict__ nb,
    const float* __restrict__ lrr, float* __restrict__ out) {
    const int lane = threadIdx.x & 31;
    const int wid  = threadIdx.x >> 5;
    const long rr  = (long)blockIdx.x * 8 + wid;   // 0 .. B*H*L-1
    const int o = (int)(rr & 8191);
    const int h = (int)((rr >> 13) & 15);
    const int b = (int)(rr >> 17);
    const int s = src_pos(o);
    const int d0 = lane * 2;
    const int colbase = h * HD_ + d0;
    const size_t yrow = ((size_t)b * L_ + s) * (size_t)NBIG;

    float2 q = *(const float2*)(Y + yrow +        colbase);
    float2 k = *(const float2*)(Y + yrow + 1024 + colbase);
    float2 v = *(const float2*)(Y + yrow + 2048 + colbase);
    float2 u = *(const float2*)(Y + yrow + 3072 + colbase);
    float2 t;
    t = *(const float2*)(qb + colbase); q.x += t.x; q.y += t.y;
    t = *(const float2*)(kb + colbase); k.x += t.x; k.y += t.y;
    t = *(const float2*)(vb + colbase); v.x += t.x; v.y += t.y;
    t = *(const float2*)(ub + colbase); u.x += t.x; u.y += t.y;

    // l2norm over HD=64 (wave reduce)
    float iq = 1.f / fmaxf(sqrtf(wave_sum(q.x * q.x + q.y * q.y)), 1e-12f);
    float ik = 1.f / fmaxf(sqrtf(wave_sum(k.x * k.x + k.y * k.y)), 1e-12f);
    float iu = 1.f / fmaxf(sqrtf(wave_sum(u.x * u.x + u.y * u.y)), 1e-12f);
    q.x *= iq; q.y *= iq;  k.x *= ik; k.y *= ik;  u.x *= iu; u.y *= iu;

    // RoPE (pair per lane, freq index = lane, indexed by SOURCE position s)
    float2 cs = *(const float2*)(freqs + ((size_t)s * 32 + lane) * 2);
    float2 qo = make_float2(q.x * cs.x - q.y * cs.y, q.x * cs.y + q.y * cs.x);
    float2 ko = make_float2(k.x * cs.x - k.y * cs.y, k.x * cs.y + k.y * cs.x);
    float2 uo = make_float2(u.x * cs.x - u.y * cs.y, u.x * cs.y + u.y * cs.x);

    // V layernorm on diff = V - rope(K), ddof=1
    float2 df = make_float2(v.x - ko.x, v.y - ko.y);
    float mean = wave_sum(df.x + df.y) * (1.0f / 64.0f);
    float sq   = wave_sum(df.x * df.x + df.y * df.y);
    float var  = (sq - 64.0f * mean * mean) * (1.0f / 63.0f);
    float istd = 1.0f / (sqrtf(fmaxf(var, 0.0f)) + 1e-8f);
    float2 w2 = *(const float2*)(nw + colbase);
    float2 b2 = *(const float2*)(nb + colbase);
    float2 vo = make_float2(w2.x * (df.x - mean) * istd + b2.x + ko.x,
                            w2.y * (df.y - mean) * istd + b2.y + ko.y);

    // eta[b,h,o_nc,o_i,j] = sigmoid(lr_raw[b, chunk(s)*64 + j, h]) / (HD*MINI)
    const int cbase = (s >> 6) << 6;
    const size_t lb = ((size_t)b * L_ + cbase) * (size_t)H_ + h;
    float r0 = lrr[lb + (size_t)d0 * H_];
    float r1 = lrr[lb + (size_t)(d0 + 1) * H_];
    float2 eo = make_float2((1.0f / (1.0f + expf(-r0))) * (1.0f / 4096.0f),
                            (1.0f / (1.0f + expf(-r1))) * (1.0f / 4096.0f));

    const int nc = o >> 6, ii = o & 63;
    const size_t tstride = (size_t)B_ * H_ * NC_ * MINI_ * HD_;
    size_t base = ((((size_t)b * H_ + h) * NC_ + nc) * (size_t)MINI_ + ii) * HD_ + d0;
    *(float2*)(out + base)               = qo;
    *(float2*)(out + base +     tstride) = ko;
    *(float2*)(out + base + 2 * tstride) = vo;
    *(float2*)(out + base + 3 * tstride) = uo;
    *(float2*)(out + base + 4 * tstride) = eo;
}

// ---------------------------------------------------------------------------
extern "C" void kernel_launch(void* const* d_in, const int* in_sizes, int n_in,
                              void* d_out, int out_size, void* d_ws, size_t ws_size,
                              hipStream_t stream) {
    const float* X   = (const float*)d_in[0];
    const float* fr  = (const float*)d_in[1];
    const float* wq  = (const float*)d_in[2];
    const float* wqb = (const float*)d_in[3];
    const float* wk  = (const float*)d_in[4];
    const float* wkb = (const float*)d_in[5];
    const float* wv  = (const float*)d_in[6];
    const float* wvb = (const float*)d_in[7];
    const float* wu  = (const float*)d_in[8];
    const float* wub = (const float*)d_in[9];
    const float* nw  = (const float*)d_in[10];
    const float* nb  = (const float*)d_in[11];
    const float* lrw = (const float*)d_in[12];
    const float* lrb = (const float*)d_in[13];
    float* out = (float*)d_out;

    char* ws = (char*)d_ws;
    size_t off = 0;
    bf16* Xb = (bf16*)(ws + off);  off += (size_t)ROWS * D_ * sizeof(bf16);     // 32 MB
    bf16* WT = (bf16*)(ws + off);  off += (size_t)NBIG * D_ * sizeof(bf16);     //  8 MB
    float* Y = (float*)(ws + off); off += (size_t)ROWS * NBIG * sizeof(float);  // 256 MB
    float* LR = (float*)(ws + off);                                             //  1 MB

    cvt_x_kernel<<<(ROWS * D_) / (256 * 8), 256, 0, stream>>>(X, Xb);
    cvt_w_kernel<<<(NBIG * D_) / 256, 256, 0, stream>>>(wq, wk, wv, wu, WT);

    dim3 gg(NBIG / BN, ROWS / BM);   // (16, 128)
    gemm_wmma_bf16<<<gg, 256, 0, stream>>>(Xb, WT, Y);

    lr_raw_kernel<<<ROWS, 256, 0, stream>>>(X, lrw, lrb, LR);

    epilogue_kernel<<<(B_ * H_ * L_) / 8, 256, 0, stream>>>(
        Y, fr, wqb, wkb, wvb, wub, nw, nb, LR, out);
}
